// MultiHeadAttention_4088808865984
// MI455X (gfx1250) — matrix operand order repaired and checked by reading, not compiled, not run on hardware
//
#include <hip/hip_runtime.h>
#include <hip/hip_bf16.h>

#define B_ 2
#define S_ 2048
#define E_ 1024
#define H_ 16
#define D_ 64

typedef _Float16 v4h  __attribute__((ext_vector_type(4)));
typedef _Float16 v8h  __attribute__((ext_vector_type(8)));
typedef _Float16 v16h __attribute__((ext_vector_type(16)));
typedef float    v8f  __attribute__((ext_vector_type(8)));

static __device__ __forceinline__ v16h cat8(v8h lo, v8h hi) {
  return __builtin_shufflevector(lo, hi, 0,1,2,3,4,5,6,7,8,9,10,11,12,13,14,15);
}
static __device__ __forceinline__ v8f wmma32(v16h a, v16h b, v8f c) {
  // D = A(16x32 f16) * B(32x16 f16) + C(16x16 f32)
  return __builtin_amdgcn_wmma_f32_16x16x32_f16(false, a, false, b, (short)0, c, false, false);
}
static __device__ __forceinline__ v8h shfl_xor16_h8(v8h x) {
  union { v8h h; int i[4]; } u; u.h = x;
  u.i[0] = __shfl_xor(u.i[0], 16, 32);
  u.i[1] = __shfl_xor(u.i[1], 16, 32);
  u.i[2] = __shfl_xor(u.i[2], 16, 32);
  u.i[3] = __shfl_xor(u.i[3], 16, 32);
  return u.h;
}

// ---------------------------------------------------------------- fp32 -> fp16
__global__ __launch_bounds__(256) void cvt_f32_f16(const float* __restrict__ s,
                                                   _Float16* __restrict__ d, int n4) {
  int i = blockIdx.x * blockDim.x + threadIdx.x;
  if (i < n4) {
    float4 f = ((const float4*)s)[i];
    v4h h;
    h[0] = (_Float16)f.x; h[1] = (_Float16)f.y;
    h[2] = (_Float16)f.z; h[3] = (_Float16)f.w;
    ((v4h*)d)[i] = h;
  }
}

// ------------------------------------------------- Y = A(MxK) * W(NxK)^T + bias
// One wave per 32x64 output tile (2 A-frags reuse 4 B-frags -> 12 b128 / 8 wmma).
// Output layout by mode:
//   0: Qh (B,H,S,D) fp16, scaled        1: Kh (B,H,S,D) fp16
//   2: V^T (B,H,D,S) fp16               3: fp32 row-major (B*S, E)
__global__ __launch_bounds__(32) void gemm_wmma(const _Float16* __restrict__ A,
                                                const _Float16* __restrict__ W,
                                                const float* __restrict__ bias,
                                                void* __restrict__ out,
                                                float scale, int mode) {
  const int lane = threadIdx.x;
  const int g    = lane >> 4;
  const int n0   = lane & 15;
  const int col0 = blockIdx.x * 64;
  const int rowb = blockIdx.y * 32;
  const size_t K = E_;
  const _Float16* arow0 = A + (size_t)(rowb + n0) * K;
  const _Float16* arow1 = A + (size_t)(rowb + 16 + n0) * K;

  v8f acc[2][4];
#pragma unroll
  for (int rr = 0; rr < 2; ++rr)
#pragma unroll
    for (int t = 0; t < 4; ++t) acc[rr][t] = (v8f){};

  for (int k0 = 0; k0 < E_; k0 += 32) {
    v16h a0 = cat8(*(const v8h*)(arow0 + k0 + 16 * g),
                   *(const v8h*)(arow0 + k0 + 8 + 16 * g));
    v16h a1 = cat8(*(const v8h*)(arow1 + k0 + 16 * g),
                   *(const v8h*)(arow1 + k0 + 8 + 16 * g));
    const _Float16* wbase = W + k0 + 16 * g;
    v16h b0 = *(const v16h*)(wbase + (size_t)(col0 +      n0) * K);
    v16h b1 = *(const v16h*)(wbase + (size_t)(col0 + 16 + n0) * K);
    v16h b2 = *(const v16h*)(wbase + (size_t)(col0 + 32 + n0) * K);
    v16h b3 = *(const v16h*)(wbase + (size_t)(col0 + 48 + n0) * K);
    acc[0][0] = wmma32(a0, b0, acc[0][0]);
    acc[0][1] = wmma32(a0, b1, acc[0][1]);
    acc[0][2] = wmma32(a0, b2, acc[0][2]);
    acc[0][3] = wmma32(a0, b3, acc[0][3]);
    acc[1][0] = wmma32(a1, b0, acc[1][0]);
    acc[1][1] = wmma32(a1, b1, acc[1][1]);
    acc[1][2] = wmma32(a1, b2, acc[1][2]);
    acc[1][3] = wmma32(a1, b3, acc[1][3]);
  }

#pragma unroll
  for (int rr = 0; rr < 2; ++rr) {
    const int rowbase = rowb + 16 * rr;
    const int bidx  = rowbase >> 11;                 // row / S
    const int srow0 = (rowbase & (S_ - 1)) + 8 * g;  // base s for this half-wave
#pragma unroll
    for (int t = 0; t < 4; ++t) {
      const int colt = col0 + 16 * t + n0;
      const float bv = bias[colt];
      if (mode <= 1) {                         // (B,H,S,D) fp16
        const int hh = colt >> 6, dd = colt & 63;
        _Float16* o = (_Float16*)out;
        size_t base = (((size_t)bidx * H_ + hh) * S_ + srow0) * D_ + dd;
#pragma unroll
        for (int r = 0; r < 8; ++r)
          o[base + (size_t)r * D_] = (_Float16)((acc[rr][t][r] + bv) * scale);
      } else if (mode == 2) {                  // V^T (B,H,D,S) fp16, packed store
        const int hh = colt >> 6, dd = colt & 63;
        _Float16* o = (_Float16*)out;
        size_t base = (((size_t)bidx * H_ + hh) * D_ + dd) * S_ + srow0;
        v8h pk;
#pragma unroll
        for (int r = 0; r < 8; ++r) pk[r] = (_Float16)(acc[rr][t][r] + bv);
        *(v8h*)(o + base) = pk;
      } else {                                 // fp32 row-major
        float* o = (float*)out;
        size_t base = (size_t)(rowbase + 8 * g) * E_ + colt;
#pragma unroll
        for (int r = 0; r < 8; ++r)
          o[base + (size_t)r * E_] = acc[rr][t][r] + bv;
      }
    }
  }
}

// --------------------------------------------------------- flash attention
// Block = 4 waves = 64 query rows of one (b,h); the 32-key K tile and V^T tile
// are staged once per block into double-buffered LDS and shared by all waves.
// Per wave: S^T = K * Q^T so each lane owns one query row (lane&15): row stats
// reduce within-lane + one shfl_xor(16). O^T = V^T * P^T accumulated in f32.
__global__ __launch_bounds__(128) void flash_attn(const _Float16* __restrict__ Qh,
                                                  const _Float16* __restrict__ Kh,
                                                  const _Float16* __restrict__ VT,
                                                  const int* __restrict__ mask,
                                                  _Float16* __restrict__ Ah) {
  __shared__ __align__(32) _Float16 Kt[2][32][64];  // [buf][key][d]
  __shared__ __align__(32) _Float16 Vt[2][64][32];  // [buf][d][key]

  const int tid  = threadIdx.x;
  const int lane = tid & 31;
  const int wv   = tid >> 5;
  const int g    = lane >> 4;
  const int n0   = lane & 15;
  const int bh   = blockIdx.x;
  const int b    = bh >> 4;   // H=16
  const int h    = bh & 15;
  const int qtile = blockIdx.y * 64 + wv * 16;

  const _Float16* Qp = Qh + (size_t)bh * S_ * D_;
  const _Float16* Kp = Kh + (size_t)bh * S_ * D_;
  const _Float16* Vp = VT + (size_t)bh * D_ * S_;
  const int* mp = mask + b * S_;

  // cooperative stage offsets: K tile 4KB, V^T tile 4KB, 128 threads x 64B
  const int krow = tid >> 2, kch = (tid & 3) * 16;   // K: 32 rows x 64 halves
  const int vrow = tid >> 1, vch = (tid & 1) * 16;   // V: 64 rows x 32 halves

  // Q^T B-fragments (reused for every key block): b[i]=Q[qtile+n0][d]
  const _Float16* qrow = Qp + (size_t)(qtile + n0) * D_;
  const v16h qb0 = cat8(*(const v8h*)(qrow + 8 * g), *(const v8h*)(qrow + 16 + 8 * g));
  const v16h qb1 = cat8(*(const v8h*)(qrow + 32 + 8 * g), *(const v8h*)(qrow + 48 + 8 * g));

  v8f acc0 = {}, acc1 = {}, acc2 = {}, acc3 = {};
  float m_run = -1e30f, l_run = 0.0f;

  // prologue: stage first tile
  *(v16h*)&Kt[0][krow][kch] = *(const v16h*)(Kp + (size_t)krow * D_ + kch);
  *(v16h*)&Vt[0][vrow][vch] = *(const v16h*)(Vp + (size_t)vrow * S_ + vch);
  __syncthreads();

  for (int kb = 0; kb < S_; kb += 32) {
    const int cur = (kb >> 5) & 1;
    if (kb + 32 < S_) {  // stage next tile into the other buffer
      const int nxt = cur ^ 1;
      *(v16h*)&Kt[nxt][krow][kch] =
          *(const v16h*)(Kp + (size_t)(kb + 32 + krow) * D_ + kch);
      *(v16h*)&Vt[nxt][vrow][vch] =
          *(const v16h*)(Vp + (size_t)vrow * S_ + kb + 32 + vch);
    }
    if (kb + 64 < S_) {  // warm L2 two blocks ahead
      __builtin_prefetch(Kp + (size_t)(kb + 64 + krow) * D_ + kch, 0, 1);
      __builtin_prefetch(Vp + (size_t)vrow * S_ + kb + 64 + vch, 0, 1);
    }

    // K A-fragments from LDS: rows = keys, K-dim = d (two steps of 32)
    const _Float16* kr0 = &Kt[cur][n0][0];
    const _Float16* kr1 = &Kt[cur][16 + n0][0];
    v16h ka00 = cat8(*(const v8h*)(kr0 +      8 * g), *(const v8h*)(kr0 + 16 + 8 * g));
    v16h ka01 = cat8(*(const v8h*)(kr0 + 32 + 8 * g), *(const v8h*)(kr0 + 48 + 8 * g));
    v16h ka10 = cat8(*(const v8h*)(kr1 +      8 * g), *(const v8h*)(kr1 + 16 + 8 * g));
    v16h ka11 = cat8(*(const v8h*)(kr1 + 32 + 8 * g), *(const v8h*)(kr1 + 48 + 8 * g));

    v8f st0 = {}, st1 = {};
    st0 = wmma32(ka00, qb0, st0); st0 = wmma32(ka01, qb1, st0); // keys kb+8g+r
    st1 = wmma32(ka10, qb0, st1); st1 = wmma32(ka11, qb1, st1); // keys kb+16+8g+r

    // mask (key dimension): lane's keys are consecutive -> int4 loads
    const int4* m4 = (const int4*)(mp + kb);
    int4 ma = m4[2 * g], mb = m4[2 * g + 1];
    int4 mc = m4[4 + 2 * g], md = m4[5 + 2 * g];
    st0[0] = ma.x ? st0[0] : -1e30f; st0[1] = ma.y ? st0[1] : -1e30f;
    st0[2] = ma.z ? st0[2] : -1e30f; st0[3] = ma.w ? st0[3] : -1e30f;
    st0[4] = mb.x ? st0[4] : -1e30f; st0[5] = mb.y ? st0[5] : -1e30f;
    st0[6] = mb.z ? st0[6] : -1e30f; st0[7] = mb.w ? st0[7] : -1e30f;
    st1[0] = mc.x ? st1[0] : -1e30f; st1[1] = mc.y ? st1[1] : -1e30f;
    st1[2] = mc.z ? st1[2] : -1e30f; st1[3] = mc.w ? st1[3] : -1e30f;
    st1[4] = md.x ? st1[4] : -1e30f; st1[5] = md.y ? st1[5] : -1e30f;
    st1[6] = md.z ? st1[6] : -1e30f; st1[7] = md.w ? st1[7] : -1e30f;

    // online softmax: row (=query) stats, within-lane + one cross-half shuffle
    float mx = fmaxf(st0[0], st1[0]);
#pragma unroll
    for (int r = 1; r < 8; ++r) mx = fmaxf(mx, fmaxf(st0[r], st1[r]));
    mx = fmaxf(mx, __shfl_xor(mx, 16, 32));
    const float m_new = fmaxf(m_run, mx);
    const float alpha = __expf(m_run - m_new);

    v8h e0, e1;
    float ps = 0.0f;
#pragma unroll
    for (int r = 0; r < 8; ++r) {
      float t0 = __expf(st0[r] - m_new);
      float t1 = __expf(st1[r] - m_new);
      ps += t0 + t1;
      e0[r] = (_Float16)t0;
      e1[r] = (_Float16)t1;
    }
    ps += __shfl_xor(ps, 16, 32);
    l_run = l_run * alpha + ps;
    m_run = m_new;

    // Build P^T B-fragment: b[i] = P[key=8g+i (i<8), 16+8g+(i-8) (i>=8)][q=n0]
    v8h x0 = shfl_xor16_h8(e0);
    v8h x1 = shfl_xor16_h8(e1);
    v16h pb = cat8(e0, e1);

    // rescale running accumulators
    acc0 *= alpha; acc1 *= alpha; acc2 *= alpha; acc3 *= alpha;

    // O^T += V^T(16x32 per d-tile) * P^T(32x16); V^T fragments from LDS
    const _Float16* vr0 = &Vt[cur][n0][0];
    const _Float16* vr1 = &Vt[cur][16 + n0][0];
    const _Float16* vr2 = &Vt[cur][32 + n0][0];
    const _Float16* vr3 = &Vt[cur][48 + n0][0];
    acc0 = wmma32(cat8(*(const v8h*)(vr0 + 8*g), *(const v8h*)(vr0 + 16 + 8*g)), pb, acc0);
    acc1 = wmma32(cat8(*(const v8h*)(vr1 + 8*g), *(const v8h*)(vr1 + 16 + 8*g)), pb, acc1);
    acc2 = wmma32(cat8(*(const v8h*)(vr2 + 8*g), *(const v8h*)(vr2 + 16 + 8*g)), pb, acc2);
    acc3 = wmma32(cat8(*(const v8h*)(vr3 + 8*g), *(const v8h*)(vr3 + 16 + 8*g)), pb, acc3);

    __syncthreads();  // cur fully consumed; nxt fully staged
  }

  // normalize and store: lane's values are d-consecutive -> packed v8h stores
  const float inv = 1.0f / l_run;
  _Float16* obase = Ah + ((size_t)b * S_ + qtile + n0) * E_ + h * D_ + 8 * g;
  v8f accs[4] = {acc0, acc1, acc2, acc3};
#pragma unroll
  for (int t = 0; t < 4; ++t) {
    v8h oh;
#pragma unroll
    for (int r = 0; r < 8; ++r) oh[r] = (_Float16)(accs[t][r] * inv);
    *(v8h*)(obase + 16 * t) = oh;
  }
}

// --------------------------------------------------------------------- launch
extern "C" void kernel_launch(void* const* d_in, const int* in_sizes, int n_in,
                              void* d_out, int out_size, void* d_ws, size_t ws_size,
                              hipStream_t stream) {
  const float* q   = (const float*)d_in[0];
  const float* k   = (const float*)d_in[1];
  const float* v   = (const float*)d_in[2];
  const int*   msk = (const int*)  d_in[3];
  const float* Wq  = (const float*)d_in[4];
  const float* bq  = (const float*)d_in[5];
  const float* Wk  = (const float*)d_in[6];
  const float* bk  = (const float*)d_in[7];
  const float* Wv  = (const float*)d_in[8];
  const float* bv  = (const float*)d_in[9];
  const float* Wo  = (const float*)d_in[10];
  const float* bo  = (const float*)d_in[11];

  const size_t QE = (size_t)B_ * S_ * E_;   // 4,194,304 halves
  const size_t EE = (size_t)E_ * E_;        // 1,048,576 halves
  _Float16* w    = (_Float16*)d_ws;
  _Float16* q16  = w;
  _Float16* k16  = q16 + QE;
  _Float16* v16_ = k16 + QE;
  _Float16* wq16 = v16_ + QE;
  _Float16* wk16 = wq16 + EE;
  _Float16* wv16 = wk16 + EE;
  _Float16* wo16 = wv16 + EE;
  _Float16* Qh   = wo16 + EE;
  _Float16* Kh   = Qh + QE;
  _Float16* VT   = Kh + QE;
  _Float16* Ah   = VT + QE;

  const int n4a = (int)(QE / 4);
  const int n4w = (int)(EE / 4);
  cvt_f32_f16<<<(n4a + 255) / 256, 256, 0, stream>>>(q,  q16,  n4a);
  cvt_f32_f16<<<(n4a + 255) / 256, 256, 0, stream>>>(k,  k16,  n4a);
  cvt_f32_f16<<<(n4a + 255) / 256, 256, 0, stream>>>(v,  v16_, n4a);
  cvt_f32_f16<<<(n4w + 255) / 256, 256, 0, stream>>>(Wq, wq16, n4w);
  cvt_f32_f16<<<(n4w + 255) / 256, 256, 0, stream>>>(Wk, wk16, n4w);
  cvt_f32_f16<<<(n4w + 255) / 256, 256, 0, stream>>>(Wv, wv16, n4w);
  cvt_f32_f16<<<(n4w + 255) / 256, 256, 0, stream>>>(Wo, wo16, n4w);

  dim3 gg(E_ / 64, (B_ * S_) / 32);
  // fold softmax scale 1/sqrt(D)=0.125 into Qh
  gemm_wmma<<<gg, 32, 0, stream>>>(q16,  wq16, bq, Qh,    0.125f, 0);
  gemm_wmma<<<gg, 32, 0, stream>>>(k16,  wk16, bk, Kh,    1.0f,   1);
  gemm_wmma<<<gg, 32, 0, stream>>>(v16_, wv16, bv, VT,    1.0f,   2);

  flash_attn<<<dim3(B_ * H_, S_ / 64), 128, 0, stream>>>(Qh, Kh, VT, msk, Ah);

  gemm_wmma<<<gg, 32, 0, stream>>>(Ah, wo16, bo, d_out, 1.0f, 3);
}
